// MultiHeadAttention_70540542869703
// MI455X (gfx1250) — compile-verified, matrix operand order checked by reading, not run on hardware
//
#include <hip/hip_runtime.h>

typedef _Float16 f16;
typedef _Float16 v4h  __attribute__((ext_vector_type(4)));
typedef _Float16 v8h  __attribute__((ext_vector_type(8)));
typedef _Float16 v16h __attribute__((ext_vector_type(16)));
typedef float    v4f  __attribute__((ext_vector_type(4)));
typedef float    v8f  __attribute__((ext_vector_type(8)));

#define BB  2
#define TT  2048
#define DD  1024
#define HH  16
#define DKK 64

// Build a 16x32 (A) or 32x16 (B) f16 WMMA operand for v_wmma_f32_16x16x32_f16.
// Per ISA layout: lane's halves 0..7 = contraction idx [sel .. sel+7],
// halves 8..15 = [sel+16 .. sel+23], sel = (lane>=16) ? 8 : 0.
// Both runs are contiguous in memory -> two global_load_b128.
__device__ __forceinline__ v16h make_ab(const f16* p0, const f16* p1) {
  v8h lo = *(const v8h*)p0;
  v8h hi = *(const v8h*)p1;
  v16h r;
#pragma unroll
  for (int i = 0; i < 8; ++i) { r[i] = lo[i]; r[i + 8] = hi[i]; }
  return r;
}

__device__ __forceinline__ v8f wmma_f16(v16h a, v16h b, v8f c) {
  return __builtin_amdgcn_wmma_f32_16x16x32_f16(false, a, false, b, (short)0, c,
                                                false, false);
}

// ---------------- Stage 1: fp32 -> f16 convert ----------------
// Source fp32 tensors are read exactly once -> non-temporal loads keep them
// out of the L2 working set. f16 outputs are re-read by GEMMs -> regular.
__global__ void cvt_f32_f16(const float* __restrict__ src, f16* __restrict__ dst,
                            int n) {
  int i = (blockIdx.x * blockDim.x + threadIdx.x) * 4;
  if (i < n) {
    v4f v = __builtin_nontemporal_load((const v4f*)(src + i));
    v4h o;
    o[0] = (f16)v[0]; o[1] = (f16)v[1]; o[2] = (f16)v[2]; o[3] = (f16)v[3];
    *(v4h*)(dst + i) = o;
  }
}

// ---------------- Stage 2/5: y = X @ W^T + bias (2x4 register blocked) -----
// X: [M][K] f16 row-major, W: [N][K] f16 row-major (torch Linear convention).
// Each wave computes a 32x64 output block = 8 WMMA tiles. Per K-step of 32:
// 12 b128 loads feed 8 v_wmma. K-loop kept at unroll 1 and B operands are
// loaded-and-consumed immediately so the scheduler cannot pile up two
// k-steps of operands and spill accumulators past the 256-VGPR window.
// mode 0: f16 out [M][N]; mode 1: f16 out transposed per batch Vt[b][col][t];
// mode 2: f32 out [M][N] (final output, never re-read -> NT stores).
__global__ void gemm_xwt(const f16* __restrict__ X, const f16* __restrict__ W,
                         const float* __restrict__ bias, void* __restrict__ out,
                         int M, int N, int K, int mode) {
  const int lane = threadIdx.x & 31;
  const int wid  = threadIdx.x >> 5;
  const int nblk = N >> 6;                 // 64-wide N blocks
  const int blkid = blockIdx.x * 4 + wid;
  const int mt2 = blkid / nblk;            // 32-tall M blocks
  const int nt4 = blkid - mt2 * nblk;
  if (mt2 >= (M >> 5)) return;

  const int sel = (lane & 16) ? 8 : 0;
  const f16* xp0 = X + (size_t)(mt2 * 32 + (lane & 15)) * K;
  const f16* xp1 = xp0 + (size_t)16 * K;
  const f16* wp[4];
#pragma unroll
  for (int j = 0; j < 4; ++j)
    wp[j] = W + (size_t)(nt4 * 64 + j * 16 + (lane & 15)) * K;

  v8f acc[2][4] = {};
#pragma unroll 1
  for (int kk = 0; kk < K; kk += 32) {
    const v16h a0 = make_ab(xp0 + kk + sel, xp0 + kk + 16 + sel);
    const v16h a1 = make_ab(xp1 + kk + sel, xp1 + kk + 16 + sel);
#pragma unroll
    for (int j = 0; j < 4; ++j) {
      const v16h bm = make_ab(wp[j] + kk + sel, wp[j] + kk + 16 + sel);
      acc[0][j] = wmma_f16(a0, bm, acc[0][j]);
      acc[1][j] = wmma_f16(a1, bm, acc[1][j]);
    }
  }

  const int mb0  = mt2 * 32 + ((lane & 16) ? 8 : 0);
  const int col0 = nt4 * 64 + (lane & 15);

#pragma unroll
  for (int j = 0; j < 4; ++j) {
    const int col = col0 + j * 16;
    const float bv = bias[col];
#pragma unroll
    for (int i = 0; i < 2; ++i) {
      const int mbase = mb0 + i * 16;
      if (mode == 2) {
        float* o = (float*)out;
#pragma unroll
        for (int r = 0; r < 8; ++r)
          __builtin_nontemporal_store(acc[i][j][r] + bv,
                                      o + (size_t)(mbase + r) * N + col);
      } else if (mode == 0) {
        f16* o = (f16*)out;
#pragma unroll
        for (int r = 0; r < 8; ++r)
          o[(size_t)(mbase + r) * N + col] = (f16)(acc[i][j][r] + bv);
      } else {  // mode 1: transposed store Vt[b][col][t]; rows r consecutive t
        f16* o = (f16*)out;
        const int b = mbase >> 11;  // 2048 rows per batch
        const int t = mbase & 2047;
        v8h pk;
#pragma unroll
        for (int r = 0; r < 8; ++r) pk[r] = (f16)(acc[i][j][r] + bv);
        *(v8h*)(o + ((size_t)b * DD + col) * TT + t) = pk;
      }
    }
  }
}

// ---------------- Stage 3: masked scaled scores (1 qtile x 4 ktiles) -------
// mask (32 MiB, [B][T][T]) is re-read once per head and scores (536 MB) are a write-once stream:
// both get non-temporal hints so Q/K stay resident in L2.
__global__ void attn_scores(const f16* __restrict__ Q, const f16* __restrict__ Km,
                            const int* __restrict__ mask,
                            float* __restrict__ attn) {
  const int lane = threadIdx.x & 31;
  const int wid  = threadIdx.x >> 5;
  const int tile = blockIdx.x * 4 + wid;  // bh * 128 qtiles * 32 ktile-groups
  const int kt4 = tile & 31;
  const int qt  = (tile >> 5) & 127;
  const int bh  = tile >> 12;
  const int b = bh >> 4, h = bh & 15;
  const int sel = (lane & 16) ? 8 : 0;

  const f16* qp = Q + ((size_t)(b * TT + (qt << 4) + (lane & 15))) * DD + h * DKK;
  const v16h a0 = make_ab(qp + sel, qp + 16 + sel);        // K-dim 0..31
  const v16h a1 = make_ab(qp + 32 + sel, qp + 48 + sel);   // K-dim 32..63

  const int qbase = (qt << 4) + ((lane & 16) ? 8 : 0);

#pragma unroll 1
  for (int kq = 0; kq < 4; ++kq) {
    const int kt = kt4 * 4 + kq;
    const f16* kp =
        Km + ((size_t)(b * TT + (kt << 4) + (lane & 15))) * DD + h * DKK;
    v8f c = {};
    c = wmma_f16(a0, make_ab(kp + sel, kp + 16 + sel), c);
    c = wmma_f16(a1, make_ab(kp + 32 + sel, kp + 48 + sel), c);

    const int kcol = (kt << 4) + (lane & 15);
    const int* mp = mask + (size_t)b * TT * TT + (size_t)qbase * TT + kcol;
    float* op = attn + ((size_t)bh * TT + qbase) * TT + kcol;
#pragma unroll
    for (int r = 0; r < 8; ++r) {
      const int mv = __builtin_nontemporal_load(mp + (size_t)r * TT);
      float s = mv ? c[r] * 0.125f : -__builtin_inff();
      __builtin_nontemporal_store(s, op + (size_t)r * TT);
    }
  }
}

// ---------------- Stage 4: row softmax (in place, pure stream -> NT) -------
__global__ void attn_softmax(float* __restrict__ attn) {
  const int lane = threadIdx.x & 31;
  const int wid  = threadIdx.x >> 5;
  const int row  = blockIdx.x * 8 + wid;  // B*H*T rows
  float* p = attn + (size_t)row * TT;

  float v[64];
  float m = -__builtin_inff();
#pragma unroll
  for (int i = 0; i < 64; ++i) {
    v[i] = __builtin_nontemporal_load(p + lane + i * 32);
    m = fmaxf(m, v[i]);
  }
#pragma unroll
  for (int off = 16; off > 0; off >>= 1) m = fmaxf(m, __shfl_xor(m, off, 32));
  float s = 0.f;
#pragma unroll
  for (int i = 0; i < 64; ++i) {
    v[i] = __expf(v[i] - m);
    s += v[i];
  }
#pragma unroll
  for (int off = 16; off > 0; off >>= 1) s += __shfl_xor(s, off, 32);
  const float inv = 1.0f / s;
#pragma unroll
  for (int i = 0; i < 64; ++i)
    __builtin_nontemporal_store(v[i] * inv, p + lane + i * 32);
}

// ---------------- Stage 5a: O = P @ V, all 4 dk-tiles per wave -------------
// The fp32 attn stream (last use) is NT-loaded & converted once, reused
// across 4 WMMAs per K-step. Vt stays L2-resident (regular loads).
__global__ void attn_pv(const float* __restrict__ attn,
                        const f16* __restrict__ Vt, f16* __restrict__ O) {
  const int lane = threadIdx.x & 31;
  const int wid  = threadIdx.x >> 5;
  const int tile = blockIdx.x * 4 + wid;  // bh * 128 qtiles
  const int qt = tile & 127;
  const int bh = tile >> 7;
  const int b = bh >> 4, h = bh & 15;
  const int sel = (lane & 16) ? 8 : 0;

  const float* ap = attn + ((size_t)bh * TT + (qt << 4) + (lane & 15)) * TT;
  const f16* vp[4];
#pragma unroll
  for (int j = 0; j < 4; ++j)
    vp[j] = Vt + ((size_t)b * DD + h * DKK + j * 16 + (lane & 15)) * TT;

  v8f acc[4] = {};
#pragma unroll 1
  for (int kk = 0; kk < TT; kk += 32) {
    v8f alo = __builtin_nontemporal_load((const v8f*)(ap + kk + sel));
    v8f ahi = __builtin_nontemporal_load((const v8f*)(ap + kk + 16 + sel));
    v16h a;
#pragma unroll
    for (int i = 0; i < 8; ++i) { a[i] = (f16)alo[i]; a[i + 8] = (f16)ahi[i]; }
#pragma unroll
    for (int j = 0; j < 4; ++j) {
      const v16h bm = make_ab(vp[j] + kk + sel, vp[j] + kk + 16 + sel);
      acc[j] = wmma_f16(a, bm, acc[j]);
    }
  }

  const int qbase = b * TT + (qt << 4) + ((lane & 16) ? 8 : 0);
#pragma unroll
  for (int j = 0; j < 4; ++j) {
    const int col = h * DKK + j * 16 + (lane & 15);
#pragma unroll
    for (int r = 0; r < 8; ++r)
      O[(size_t)(qbase + r) * DD + col] = (f16)acc[j][r];
  }
}

extern "C" void kernel_launch(void* const* d_in, const int* in_sizes, int n_in,
                              void* d_out, int out_size, void* d_ws,
                              size_t ws_size, hipStream_t stream) {
  (void)in_sizes; (void)n_in; (void)out_size; (void)ws_size;
  const float* x_q  = (const float*)d_in[0];
  const float* x_kv = (const float*)d_in[1];
  const int*   mask = (const int*)d_in[2];
  const float* Wq = (const float*)d_in[3];
  const float* bq = (const float*)d_in[4];
  const float* Wk = (const float*)d_in[5];
  const float* bk = (const float*)d_in[6];
  const float* Wv = (const float*)d_in[7];
  const float* bv = (const float*)d_in[8];
  const float* Wo = (const float*)d_in[9];
  const float* bo = (const float*)d_in[10];

  float* out  = (float*)d_out;
  float* attn = out + (size_t)BB * TT * DD;  // attn region of d_out

  const size_t XE = (size_t)BB * TT * DD;  // 4,194,304
  const size_t WE = (size_t)DD * DD;       // 1,048,576

  char* ws = (char*)d_ws;
  size_t off = 0;
  auto alloc = [&](size_t bytes) -> void* {
    void* p = ws + off;
    off += (bytes + 255) & ~(size_t)255;
    return p;
  };
  f16* xq16  = (f16*)alloc(XE * 2);
  f16* xkv16 = (f16*)alloc(XE * 2);
  f16* wq16  = (f16*)alloc(WE * 2);
  f16* wk16  = (f16*)alloc(WE * 2);
  f16* wv16  = (f16*)alloc(WE * 2);
  f16* wo16  = (f16*)alloc(WE * 2);
  f16* q16   = (f16*)alloc(XE * 2);  // [B*T][D]
  f16* k16   = (f16*)alloc(XE * 2);  // [B*T][D]
  f16* vt16  = (f16*)alloc(XE * 2);  // transposed: [B][D][T]
  f16* o16   = (f16*)alloc(XE * 2);  // heads recombined [B*T][D]

  // Stage 1: converts
  cvt_f32_f16<<<(int)(XE / 4 / 256), 256, 0, stream>>>(x_q, xq16, (int)XE);
  cvt_f32_f16<<<(int)(XE / 4 / 256), 256, 0, stream>>>(x_kv, xkv16, (int)XE);
  cvt_f32_f16<<<(int)(WE / 4 / 256), 256, 0, stream>>>(Wq, wq16, (int)WE);
  cvt_f32_f16<<<(int)(WE / 4 / 256), 256, 0, stream>>>(Wk, wk16, (int)WE);
  cvt_f32_f16<<<(int)(WE / 4 / 256), 256, 0, stream>>>(Wv, wv16, (int)WE);
  cvt_f32_f16<<<(int)(WE / 4 / 256), 256, 0, stream>>>(Wo, wo16, (int)WE);

  // Stage 2: QKV projections. (4096/32)*(1024/64)=2048 waves -> 512 blocks
  const int M = BB * TT;
  gemm_xwt<<<512, 128, 0, stream>>>(xq16, wq16, bq, q16, M, DD, DD, 0);
  gemm_xwt<<<512, 128, 0, stream>>>(xkv16, wk16, bk, k16, M, DD, DD, 0);
  gemm_xwt<<<512, 128, 0, stream>>>(xkv16, wv16, bv, vt16, M, DD, DD, 1);

  // Stage 3: scores. B*H*128*32 = 131072 waves -> 32768 blocks x 4 waves
  attn_scores<<<32768, 128, 0, stream>>>(q16, k16, mask, attn);

  // Stage 4: softmax. 65536 rows -> 8192 blocks x 8 waves
  attn_softmax<<<8192, 256, 0, stream>>>(attn);

  // Stage 5a: P @ V. B*H*128 = 4096 waves -> 1024 blocks x 4 waves
  attn_pv<<<1024, 128, 0, stream>>>(attn, vt16, o16);

  // Stage 5b: output projection (f32 -> d_out)
  gemm_xwt<<<512, 128, 0, stream>>>(o16, wo16, bo, out, M, DD, DD, 2);
}
